// TemporalAttentionLayer_36026185678966
// MI455X (gfx1250) — matrix operand order repaired and checked by reading, not compiled, not run on hardware
//
#include <hip/hip_runtime.h>
#include <hip/hip_bf16.h>
#include <math.h>

#define IN_DIM   256
#define OUT_DIM  256
#define HEADS    8
#define HEAD_DIM 32
#define NEG_SLOPE 0.2f
#define LN_EPS    1e-5f

typedef __attribute__((ext_vector_type(16))) _Float16 v16h;
typedef __attribute__((ext_vector_type(8)))  float    v8f;

// ---------- helpers: order-preserving float<->uint for atomic max ----------
__device__ __forceinline__ unsigned int fenc(float x) {
    unsigned int u = __float_as_uint(x);
    return (u & 0x80000000u) ? ~u : (u | 0x80000000u);
}
__device__ __forceinline__ float fdec(unsigned int k) {
    unsigned int u = (k & 0x80000000u) ? (k & 0x7FFFFFFFu) : ~k;
    return __uint_as_float(u);
}

// ---------- precision conversion kernels ----------
__global__ void k_convert_x(const float* __restrict__ x, _Float16* __restrict__ xh, int n) {
    int i = blockIdx.x * 256 + threadIdx.x;
    if (i < n) xh[i] = (_Float16)x[i];
}

// Wt[n][k] = W[k][n] in f16 (so B-fragment rows are contiguous per output column)
__global__ void k_convert_wT(const float* __restrict__ W, _Float16* __restrict__ Wt) {
    int i = blockIdx.x * 256 + threadIdx.x;        // 65536 elements
    int k = i >> 8, n = i & 255;
    Wt[(size_t)n * IN_DIM + k] = (_Float16)W[(size_t)k * OUT_DIM + n];
}

__global__ void k_init(float* __restrict__ out, unsigned int* __restrict__ m,
                       float* __restrict__ denom, int nOut, int nNH) {
    int i = blockIdx.x * 256 + threadIdx.x;
    if (i < nOut) out[i] = 0.f;
    if (i < nNH) { m[i] = 0x007FFFFFu; /* enc(-inf) */ denom[i] = 0.f; }
}

// ---------- WMMA GEMM: h[nNodes,256] = x @ W (f16 inputs, f32 accum) ----------
// one wave = one 16x16 output tile; K-loop: 8 x v_wmma_f32_16x16x32_f16
__global__ void __launch_bounds__(256) k_gemm(const _Float16* __restrict__ xh,
                                              const _Float16* __restrict__ Wt,
                                              float* __restrict__ h,
                                              int nNodes, int nTilesM) {
    const int lane  = threadIdx.x & 31;
    const int wave  = blockIdx.x * (blockDim.x >> 5) + (threadIdx.x >> 5);
    const int tileM = wave >> 4;            // 16 N-tiles per row-block (N=256)
    const int tileN = wave & 15;
    if (tileM >= nTilesM) return;
    const int hi  = lane >> 4;              // lane half select
    const int l15 = lane & 15;

    int arow_idx = tileM * 16 + l15;        // A row (M) held by this lane
    if (arow_idx >= nNodes) arow_idx = 0;   // clamp (tail tiles); store is guarded
    const unsigned int* arow = (const unsigned int*)(xh + (size_t)arow_idx * IN_DIM);
    const unsigned int* brow = (const unsigned int*)(Wt + (size_t)(tileN * 16 + l15) * IN_DIM);

    v8f c = {};
    #pragma unroll
    for (int k0 = 0; k0 < IN_DIM; k0 += 32) {
        union { v16h v; unsigned int u[8]; } A, B;
        const unsigned int* ap = arow + (k0 >> 1);   // pair (2 x f16) granularity
        const unsigned int* bp = brow + (k0 >> 1);
        #pragma unroll
        for (int j = 0; j < 8; ++j) {
            // A 16x32 f16 layout: lanes0-15 pairs {0-3 (K0-7), 8-11 (K16-23)},
            //                      lanes16-31 pairs {4-7 (K8-15), 12-15 (K24-31)}
            A.u[j] = ap[j + ((j >= 4) ? 4 : 0) + (hi ? 4 : 0)];
            // B 32x16 f16 layout: same K order as A
            B.u[j] = bp[j + ((j >= 4) ? 4 : 0) + (hi ? 4 : 0)];
        }
        c = __builtin_amdgcn_wmma_f32_16x16x32_f16(false, A.v, false, B.v,
                                                   (short)0, c, false, false);
    }
    // C/D layout: N = lane%16, VGPR r -> M = r (lanes0-15) / 8+r (lanes16-31)
    #pragma unroll
    for (int r = 0; r < 8; ++r) {
        int row = tileM * 16 + r + (hi ? 8 : 0);
        if (row < nNodes)
            h[(size_t)row * OUT_DIM + tileN * 16 + l15] = c[r];
    }
}

// ---------- per-node attention logits: a_src/a_dst [N,H] ----------
__global__ void k_logits(const float* __restrict__ h, const float* __restrict__ att_src,
                         const float* __restrict__ att_dst,
                         float* __restrict__ a_src, float* __restrict__ a_dst, int nNodes) {
    int t = blockIdx.x * 256 + threadIdx.x;
    if (t >= nNodes * HEADS) return;
    int n = t >> 3, hd = t & 7;
    const float* hp = h + (size_t)n * OUT_DIM + hd * HEAD_DIM;
    const float* as = att_src + hd * HEAD_DIM;
    const float* ad = att_dst + hd * HEAD_DIM;
    float s = 0.f, d = 0.f;
    #pragma unroll
    for (int i = 0; i < HEAD_DIM; ++i) { float v = hp[i]; s += v * as[i]; d += v * ad[i]; }
    a_src[t] = s; a_dst[t] = d;
}

// ---------- softmax pass 1: segment max over dst ----------
__global__ void k_edge_max(const int* __restrict__ src, const int* __restrict__ dst,
                           const float* __restrict__ a_src, const float* __restrict__ a_dst,
                           unsigned int* __restrict__ m, int NE, int nNodes) {
    int e = blockIdx.x * 256 + threadIdx.x;
    if (e >= NE + nNodes) return;
    int s = (e < NE) ? src[e] : (e - NE);   // self-loops appended
    int d = (e < NE) ? dst[e] : (e - NE);
    #pragma unroll
    for (int hd = 0; hd < HEADS; ++hd) {
        float v = a_src[s * HEADS + hd] + a_dst[d * HEADS + hd];
        v = (v > 0.f) ? v : NEG_SLOPE * v;
        atomicMax(&m[d * HEADS + hd], fenc(v));
    }
}

// ---------- softmax pass 2: segment sum of exp(e - max) ----------
__global__ void k_edge_expsum(const int* __restrict__ src, const int* __restrict__ dst,
                              const float* __restrict__ a_src, const float* __restrict__ a_dst,
                              const unsigned int* __restrict__ m, float* __restrict__ denom,
                              int NE, int nNodes) {
    int e = blockIdx.x * 256 + threadIdx.x;
    if (e >= NE + nNodes) return;
    int s = (e < NE) ? src[e] : (e - NE);
    int d = (e < NE) ? dst[e] : (e - NE);
    #pragma unroll
    for (int hd = 0; hd < HEADS; ++hd) {
        float v = a_src[s * HEADS + hd] + a_dst[d * HEADS + hd];
        v = (v > 0.f) ? v : NEG_SLOPE * v;
        atomicAdd(&denom[d * HEADS + hd], expf(v - fdec(m[d * HEADS + hd])));
    }
}

// ---------- pass 3: out[dst] += alpha * h[src]; one wave per edge ----------
__global__ void __launch_bounds__(256) k_edge_scatter(
        const int* __restrict__ src, const int* __restrict__ dst,
        const float* __restrict__ a_src, const float* __restrict__ a_dst,
        const unsigned int* __restrict__ m, const float* __restrict__ denom,
        const float* __restrict__ h, float* __restrict__ out, int NE, int nNodes) {
    int lane = threadIdx.x & 31;
    int e = blockIdx.x * 8 + (threadIdx.x >> 5);   // wave-uniform edge id
    if (e >= NE + nNodes) return;
    int s = (e < NE) ? src[e] : (e - NE);
    int d = (e < NE) ? dst[e] : (e - NE);
    const float* hrow = h   + (size_t)s * OUT_DIM;
    float*       orow = out + (size_t)d * OUT_DIM;
    __builtin_prefetch(hrow + lane * 8, 0, 3);     // global_prefetch_b8
    float alpha[HEADS];
    #pragma unroll
    for (int hd = 0; hd < HEADS; ++hd) {
        float v = a_src[s * HEADS + hd] + a_dst[d * HEADS + hd];
        v = (v > 0.f) ? v : NEG_SLOPE * v;
        alpha[hd] = expf(v - fdec(m[d * HEADS + hd])) / denom[d * HEADS + hd];
    }
    #pragma unroll
    for (int hd = 0; hd < HEADS; ++hd) {
        int c = hd * 32 + lane;                    // coalesced per head
        atomicAdd(&orow[c], alpha[hd] * hrow[c]);
    }
}

// ---------- bias + LayerNorm, one wave per node ----------
__global__ void __launch_bounds__(256) k_bias_ln(float* __restrict__ out,
        const float* __restrict__ bias, const float* __restrict__ gamma,
        const float* __restrict__ beta, int nNodes) {
    int lane = threadIdx.x & 31;
    int n = blockIdx.x * 8 + (threadIdx.x >> 5);
    if (n >= nNodes) return;
    float* row = out + (size_t)n * OUT_DIM;
    float v[8], s = 0.f, sq = 0.f;
    #pragma unroll
    for (int j = 0; j < 8; ++j) {
        int c = j * 32 + lane;
        v[j] = row[c] + bias[c];
        s += v[j]; sq += v[j] * v[j];
    }
    #pragma unroll
    for (int off = 16; off > 0; off >>= 1) {       // wave32 reduction
        s  += __shfl_xor(s,  off, 32);
        sq += __shfl_xor(sq, off, 32);
    }
    float mu  = s * (1.f / OUT_DIM);
    float var = sq * (1.f / OUT_DIM) - mu * mu;
    float inv = rsqrtf(var + LN_EPS);
    #pragma unroll
    for (int j = 0; j < 8; ++j) {
        int c = j * 32 + lane;
        row[c] = (v[j] - mu) * inv * gamma[c] + beta[c];
    }
}

extern "C" void kernel_launch(void* const* d_in, const int* in_sizes, int n_in,
                              void* d_out, int out_size, void* d_ws, size_t ws_size,
                              hipStream_t stream) {
    (void)n_in; (void)out_size; (void)ws_size;
    const float* x       = (const float*)d_in[0];
    const int*   ei      = (const int*)  d_in[1];   // per harness doc: integer -> const int*
    const float* W       = (const float*)d_in[2];
    const float* att_src = (const float*)d_in[3];
    const float* att_dst = (const float*)d_in[4];
    const float* bias    = (const float*)d_in[5];
    const float* gamma   = (const float*)d_in[6];
    const float* beta    = (const float*)d_in[7];
    float* out = (float*)d_out;

    const int nNodes = in_sizes[0] / IN_DIM;        // 30000
    const int NE     = in_sizes[1] / 2;             // 480000
    const int* src = ei;
    const int* dst = ei + NE;

    // workspace carve-out (256B aligned)
    char* ws = (char*)d_ws;
    size_t off = 0;
    auto alloc = [&](size_t bytes) -> char* {
        char* p = ws + off;
        off = (off + bytes + 255) & ~(size_t)255;
        return p;
    };
    _Float16*     xh    = (_Float16*)    alloc((size_t)nNodes * IN_DIM * sizeof(_Float16));
    _Float16*     Wt    = (_Float16*)    alloc((size_t)IN_DIM * OUT_DIM * sizeof(_Float16));
    float*        h     = (float*)       alloc((size_t)nNodes * OUT_DIM * sizeof(float));
    float*        aS    = (float*)       alloc((size_t)nNodes * HEADS * sizeof(float));
    float*        aD    = (float*)       alloc((size_t)nNodes * HEADS * sizeof(float));
    unsigned int* mMax  = (unsigned int*)alloc((size_t)nNodes * HEADS * sizeof(unsigned int));
    float*        denom = (float*)       alloc((size_t)nNodes * HEADS * sizeof(float));

    const int nX   = nNodes * IN_DIM;
    const int nOut = nNodes * OUT_DIM;
    const int nNH  = nNodes * HEADS;
    const int totE = NE + nNodes;

    k_convert_x <<<(nX + 255) / 256, 256, 0, stream>>>(x, xh, nX);
    k_convert_wT<<<(IN_DIM * OUT_DIM) / 256, 256, 0, stream>>>(W, Wt);
    k_init      <<<(nOut + 255) / 256, 256, 0, stream>>>(out, mMax, denom, nOut, nNH);

    const int nTilesM = (nNodes + 15) / 16;
    const int waves   = nTilesM * 16;               // 16 N-tiles per M-tile
    k_gemm<<<(waves + 7) / 8, 256, 0, stream>>>(xh, Wt, h, nNodes, nTilesM);

    k_logits      <<<(nNH + 255) / 256, 256, 0, stream>>>(h, att_src, att_dst, aS, aD, nNodes);
    k_edge_max    <<<(totE + 255) / 256, 256, 0, stream>>>(src, dst, aS, aD, mMax, NE, nNodes);
    k_edge_expsum <<<(totE + 255) / 256, 256, 0, stream>>>(src, dst, aS, aD, mMax, denom, NE, nNodes);
    k_edge_scatter<<<(totE + 7) / 8, 256, 0, stream>>>(src, dst, aS, aD, mMax, denom, h, out, NE, nNodes);
    k_bias_ln     <<<(nNodes + 7) / 8, 256, 0, stream>>>(out, bias, gamma, beta, nNodes);
}
